// Attention_58093727645696
// MI455X (gfx1250) — matrix operand order repaired and checked by reading, not compiled, not run on hardware
//
#include <hip/hip_runtime.h>
#include <hip/hip_bf16.h>

typedef __attribute__((ext_vector_type(16))) __bf16 v16bf;
typedef __attribute__((ext_vector_type(8)))  __bf16 v8bf;
typedef __attribute__((ext_vector_type(8)))  float  v8f;

constexpr int kS    = 2048;   // sequence length
constexpr int kHID  = 2048;   // hidden dim
constexpr int kNH   = 32;     // query heads
constexpr int kNKV  = 8;      // kv heads
constexpr int kHD   = 64;     // head dim
constexpr int kROT  = 32;     // rope half
constexpr int kWIN  = 128;    // sliding window
constexpr float kSCALE = 0.125f; // 1/sqrt(64)

// ---------------------------------------------------------------------------
// WMMA helpers (CDNA5 gfx1250, wave32)
// ---------------------------------------------------------------------------
__device__ __forceinline__ v8f wmma_bf16(v16bf a, v16bf b, v8f c) {
  return __builtin_amdgcn_wmma_f32_16x16x32_bf16(
      /*neg_a=*/false, a, /*neg_b=*/false, b,
      /*c_mod=*/(short)0, c, /*reuse_a=*/false, /*reuse_b=*/false);
}

// A fragment: 16 rows x 32 K, src row-major, leading dim ld (elements).
__device__ __forceinline__ v16bf load_A(const __bf16* base, int ld) {
  int lane = threadIdx.x & 31;
  const __bf16* p = base + (lane & 15) * ld + ((lane >> 4) << 4);
  v8bf lo = *(const v8bf*)p;
  v8bf hi = *(const v8bf*)(p + 8);
  v16bf r;
#pragma unroll
  for (int e = 0; e < 8; ++e) { r[e] = lo[e]; r[e + 8] = hi[e]; }
  return r;
}

// B fragment: 32 K x 16 N with B[k][n] = W[n][k], W row-major, ld (elements).
__device__ __forceinline__ v16bf load_B(const __bf16* base, int ld) {
  int lane = threadIdx.x & 31;
  return *(const v16bf*)(base + (lane & 15) * ld + ((lane >> 4) << 4));
}

// ---------------------------------------------------------------------------
// CDNA5 async copy: 16 bytes per lane, global -> LDS, tracked by ASYNCcnt.
// ---------------------------------------------------------------------------
__device__ __forceinline__ void async_cp16(const __bf16* gbase, unsigned goff,
                                           const __bf16* ldst) {
  unsigned l = (unsigned)(size_t)ldst;   // low 32 bits of generic ptr = LDS offset
  asm volatile("global_load_async_to_lds_b128 %0, %1, %2 offset:0"
               :: "v"(l), "v"(goff), "s"(gbase) : "memory");
}

__device__ __forceinline__ unsigned panel_goff(int c, int k) {
  // chunk c of a row-major panel with ld = kHID bf16: row = c>>2, 16B piece = c&3
  return ((unsigned)(c >> 2) * (kHID * 2u)) + ((unsigned)(c & 3) * 16u) +
         (unsigned)(k * 2);
}

// ---------------------------------------------------------------------------
// 4-stage pipelined GEMM mainloop, K=2048 in 64 steps of 32.
// Workgroup tile: 64 rows(X) x 128 cols(W); wave = 16 rows x 128 cols, 8 acc.
// LDS: sX[4][64*32], sW[4][128*32]  (48 KB).  3 async batches in flight
// (6 instrs each).  One barrier per k-step: wait -> barrier -> issue -> math.
// B fragments preloaded in groups of 4 so DS waits can be partial.
// ---------------------------------------------------------------------------
__device__ __forceinline__ void gemm_mainloop(const __bf16* gX, const __bf16* gW,
                                              __bf16* sX, __bf16* sW, v8f* acc) {
  const int wave = threadIdx.x >> 5;
  const int lane = threadIdx.x & 31;
  constexpr int STEPS = kHID / 32;

  auto issue = [&](int stage, int k) {
    __bf16* dX = sX + stage * (64 * 32);
    __bf16* dW = sW + stage * (128 * 32);
#pragma unroll
    for (int i = 0; i < 2; ++i) {          // X panel: 256 chunks / 4 waves
      int c = wave * 64 + i * 32 + lane;
      async_cp16(gX, panel_goff(c, k), dX + c * 8);
    }
#pragma unroll
    for (int i = 0; i < 4; ++i) {          // W panel: 512 chunks / 4 waves
      int c = wave * 128 + i * 32 + lane;
      async_cp16(gW, panel_goff(c, k), dW + c * 8);
    }
  };

  issue(0, 0);
  issue(1, 32);
  issue(2, 64);

  for (int kk4 = 0; kk4 < STEPS; kk4 += 4) {
#pragma unroll
    for (int u = 0; u < 4; ++u) {          // stage index is compile-time = u
      const int kk = kk4 + u;
      if (kk + 2 < STEPS)      asm volatile("s_wait_asynccnt 12" ::: "memory");
      else if (kk + 1 < STEPS) asm volatile("s_wait_asynccnt 6"  ::: "memory");
      else                     asm volatile("s_wait_asynccnt 0"  ::: "memory");
      __syncthreads();   // publishes stage u to all waves; retires old readers
      if (kk + 3 < STEPS) issue((u + 3) & 3, (kk + 3) * 32);
      v16bf a = load_A(sX + u * (64 * 32) + wave * 16 * 32, 32);
#pragma unroll
      for (int g = 0; g < 2; ++g) {        // batch 4 B frags -> partial ds waits
        v16bf b0 = load_B(sW + u * (128 * 32) + (g * 4 + 0) * 16 * 32, 32);
        v16bf b1 = load_B(sW + u * (128 * 32) + (g * 4 + 1) * 16 * 32, 32);
        v16bf b2 = load_B(sW + u * (128 * 32) + (g * 4 + 2) * 16 * 32, 32);
        v16bf b3 = load_B(sW + u * (128 * 32) + (g * 4 + 3) * 16 * 32, 32);
        acc[g * 4 + 0] = wmma_bf16(a, b0, acc[g * 4 + 0]);
        acc[g * 4 + 1] = wmma_bf16(a, b1, acc[g * 4 + 1]);
        acc[g * 4 + 2] = wmma_bf16(a, b2, acc[g * 4 + 2]);
        acc[g * 4 + 3] = wmma_bf16(a, b3, acc[g * 4 + 3]);
      }
    }
  }
}

// ---------------------------------------------------------------------------
// fp32 -> bf16 conversion
// ---------------------------------------------------------------------------
__global__ void cvt_bf16_kernel(const float* __restrict__ src,
                                __bf16* __restrict__ dst, int n) {
  int i = blockIdx.x * blockDim.x + threadIdx.x;
  if (i < n) dst[i] = (__bf16)src[i];
}

// ---------------------------------------------------------------------------
// Fused QKV projection + bias + RoPE.  128-col blocks:
// [0,2048) -> Q, [2048,2560) -> K, [2560,3072) -> V; each block = 2 heads.
// ---------------------------------------------------------------------------
__global__ void qkv_kernel(const __bf16* __restrict__ xb,
                           const __bf16* __restrict__ wqb,
                           const __bf16* __restrict__ wkb,
                           const __bf16* __restrict__ wvb,
                           const float*  __restrict__ bq,
                           const float*  __restrict__ bk,
                           const float*  __restrict__ bv,
                           const float*  __restrict__ cosT,
                           const float*  __restrict__ sinT,
                           __bf16* __restrict__ qbuf,    // [S][2048]
                           __bf16* __restrict__ kbuf,    // [NKV][S][64]
                           __bf16* __restrict__ vtbuf) { // [NKV][64][S]
  __shared__ __align__(16) __bf16 sX[4][64 * 32];
  __shared__ __align__(16) __bf16 sW[4][128 * 32];

  const int sb   = blockIdx.x * 64;
  const int cb   = blockIdx.y * 128;
  const int wave = threadIdx.x >> 5;
  const int lane = threadIdx.x & 31;
  const int hi   = lane >> 4;
  const int lo   = lane & 15;
  const int s0   = sb + wave * 16;

  const __bf16* wbase;
  const float*  bias;
  int wrow0;
  if (cb < kHID)            { wbase = wqb; wrow0 = cb;              bias = bq + wrow0; }
  else if (cb < kHID + 512) { wbase = wkb; wrow0 = cb - kHID;       bias = bk + wrow0; }
  else                      { wbase = wvb; wrow0 = cb - kHID - 512; bias = bv + wrow0; }

  v8f acc[8] = {};
  gemm_mainloop(xb + sb * kHID, wbase + wrow0 * kHID, &sX[0][0], &sW[0][0], acc);

  // bias
#pragma unroll
  for (int t = 0; t < 8; ++t) {
    float bval = bias[t * 16 + lo];
#pragma unroll
    for (int r = 0; r < 8; ++r) acc[t][r] += bval;
  }

  // RoPE (Q and K): within each head-half (tiles th..th+3), pair (t, t+2).
  if (cb < kHID + 512) {
#pragma unroll
    for (int th = 0; th < 8; th += 4)
#pragma unroll
      for (int tt = 0; tt < 2; ++tt) {
        int t = th + tt;
#pragma unroll
        for (int r = 0; r < 8; ++r) {
          int srow = s0 + r + 8 * hi;
          int dd   = tt * 16 + lo;                  // 0..31 within head
          float c  = cosT[srow * kROT + dd];
          float sn = sinT[srow * kROT + dd];
          float x1 = acc[t][r], x2 = acc[t + 2][r];
          acc[t][r]     = x1 * c - x2 * sn;
          acc[t + 2][r] = x1 * sn + x2 * c;
        }
      }
  }

  // stores
  if (cb < kHID) {
#pragma unroll
    for (int t = 0; t < 8; ++t)
#pragma unroll
      for (int r = 0; r < 8; ++r) {
        int srow = s0 + r + 8 * hi;
        qbuf[srow * kHID + cb + t * 16 + lo] = (__bf16)acc[t][r];
      }
  } else if (cb < kHID + 512) {
    int base = cb - kHID;
#pragma unroll
    for (int t = 0; t < 8; ++t) {
      int hk = (base >> 6) + (t >> 2);
      int dl = (t & 3) * 16 + lo;
#pragma unroll
      for (int r = 0; r < 8; ++r) {
        int srow = s0 + r + 8 * hi;
        kbuf[(hk * kS + srow) * kHD + dl] = (__bf16)acc[t][r];
      }
    }
  } else {
    int base = cb - kHID - 512;
#pragma unroll
    for (int t = 0; t < 8; ++t) {
      int hk = (base >> 6) + (t >> 2);
      int dl = (t & 3) * 16 + lo;
#pragma unroll
      for (int r = 0; r < 8; ++r) {
        int srow = s0 + r + 8 * hi;
        vtbuf[(hk * kHD + dl) * kS + srow] = (__bf16)acc[t][r];
      }
    }
  }
}

// ---------------------------------------------------------------------------
// Sliding-window flash attention with sink logit.
// grid = (heads=32, qblocks=32); block = 128 (4 independent waves, 16 rows each).
// ---------------------------------------------------------------------------
__global__ void attn_kernel(const __bf16* __restrict__ qbuf,
                            const __bf16* __restrict__ kbuf,
                            const __bf16* __restrict__ vtbuf,
                            const float*  __restrict__ sinks,
                            __bf16* __restrict__ obuf) { // [S][2048]
  __shared__ __align__(32) __bf16 pls[4][16 * 32]; // per-wave P (16x32) buffer

  const int h    = blockIdx.x;
  const int q0   = blockIdx.y * 64;
  const int wave = threadIdx.x >> 5;
  const int lane = threadIdx.x & 31;
  const int hi   = lane >> 4;
  const int lo   = lane & 15;
  const int qw   = q0 + wave * 16;
  const int hk   = h >> 2;                  // n_rep = 4

  v16bf qa0 = load_A(qbuf + qw * kHID + h * kHD, kHID);
  v16bf qa1 = load_A(qbuf + qw * kHID + h * kHD + 32, kHID);

  v8f o[4] = {};
  float m[8], l[8], alpha[8];
  const float sk = sinks[h];
#pragma unroll
  for (int r = 0; r < 8; ++r) { m[r] = sk; l[r] = 1.0f; } // sink folded in

  int jbs = qw - kWIN; if (jbs < 0) jbs = 0; jbs &= ~31;
  __bf16* pl = &pls[wave][0];

  for (int jb = jbs; jb <= qw + 15; jb += 32) {
    v8f sc[2];
#pragma unroll
    for (int t = 0; t < 2; ++t) {
      const __bf16* kb = kbuf + (hk * kS + jb + t * 16) * kHD;
      v16bf b0 = load_B(kb, kHD);
      v16bf b1 = load_B(kb + 32, kHD);
      v8f z = {};
      z = wmma_bf16(qa0, b0, z);
      z = wmma_bf16(qa1, b1, z);
      sc[t] = z;
    }
#pragma unroll
    for (int t = 0; t < 2; ++t)
#pragma unroll
      for (int r = 0; r < 8; ++r) {
        int i = qw + r + 8 * hi;
        int j = jb + t * 16 + lo;
        bool ok = (i >= j) && (i - j <= kWIN);
        sc[t][r] = ok ? sc[t][r] * kSCALE : -1e30f;
      }
#pragma unroll
    for (int r = 0; r < 8; ++r) {
      float tm = fmaxf(sc[0][r], sc[1][r]);
      tm = fmaxf(tm, __shfl_xor(tm, 1));
      tm = fmaxf(tm, __shfl_xor(tm, 2));
      tm = fmaxf(tm, __shfl_xor(tm, 4));
      tm = fmaxf(tm, __shfl_xor(tm, 8));
      float mn = fmaxf(m[r], tm);
      float a  = __expf(m[r] - mn);
      float e0 = __expf(sc[0][r] - mn);
      float e1 = __expf(sc[1][r] - mn);
      float rs = e0 + e1;
      rs += __shfl_xor(rs, 1);
      rs += __shfl_xor(rs, 2);
      rs += __shfl_xor(rs, 4);
      rs += __shfl_xor(rs, 8);
      m[r] = mn; l[r] = l[r] * a + rs; alpha[r] = a;
      sc[0][r] = e0; sc[1][r] = e1;
    }
#pragma unroll
    for (int dt = 0; dt < 4; ++dt)
#pragma unroll
      for (int r = 0; r < 8; ++r) o[dt][r] *= alpha[r];

    // P: C-layout -> A-layout via per-wave LDS bounce (wave-local, in-order)
#pragma unroll
    for (int t = 0; t < 2; ++t)
#pragma unroll
      for (int r = 0; r < 8; ++r)
        pl[(r + 8 * hi) * 32 + t * 16 + lo] = (__bf16)sc[t][r];
    asm volatile("s_wait_dscnt 0" ::: "memory");
    v16bf pa = load_A(pl, 32);

#pragma unroll
    for (int dt = 0; dt < 4; ++dt) {
      v16bf vb = load_B(vtbuf + (hk * kHD + dt * 16) * kS + jb, kS);
      o[dt] = wmma_bf16(pa, vb, o[dt]);
    }
  }

#pragma unroll
  for (int dt = 0; dt < 4; ++dt)
#pragma unroll
    for (int r = 0; r < 8; ++r) {
      int srow = qw + r + 8 * hi;
      obuf[srow * kHID + h * kHD + dt * 16 + lo] = (__bf16)(o[dt][r] / l[r]);
    }
}

// ---------------------------------------------------------------------------
// Output projection: out[s,c] = sum_o attn[s,o] * wo[c,o] + bo[c]   (fp32 out)
// ---------------------------------------------------------------------------
__global__ void oproj_kernel(const __bf16* __restrict__ obuf,
                             const __bf16* __restrict__ wob,
                             const float*  __restrict__ bo,
                             float* __restrict__ out) {
  __shared__ __align__(16) __bf16 sX[4][64 * 32];
  __shared__ __align__(16) __bf16 sW[4][128 * 32];

  const int sb   = blockIdx.x * 64;
  const int cb   = blockIdx.y * 128;
  const int wave = threadIdx.x >> 5;
  const int lane = threadIdx.x & 31;
  const int hi   = lane >> 4;
  const int lo   = lane & 15;
  const int s0   = sb + wave * 16;

  v8f acc[8] = {};
  gemm_mainloop(obuf + sb * kHID, wob + cb * kHID, &sX[0][0], &sW[0][0], acc);

#pragma unroll
  for (int t = 0; t < 8; ++t) {
    float bval = bo[cb + t * 16 + lo];
#pragma unroll
    for (int r = 0; r < 8; ++r) {
      int srow = s0 + r + 8 * hi;
      out[srow * kHID + cb + t * 16 + lo] = acc[t][r] + bval;
    }
  }
}

// ---------------------------------------------------------------------------
// Host launcher.  Workspace layout (bytes):
//   xb 8M | wqb 8M | wkb 2M | wvb 2M | wob 8M | qb 8M | kb 2M | vt 2M | ob 8M
// ---------------------------------------------------------------------------
extern "C" void kernel_launch(void* const* d_in, const int* in_sizes, int n_in,
                              void* d_out, int out_size, void* d_ws, size_t ws_size,
                              hipStream_t stream) {
  const float* x    = (const float*)d_in[0];
  const float* cosT = (const float*)d_in[1];
  const float* sinT = (const float*)d_in[2];
  const float* wq_w = (const float*)d_in[3];
  const float* wq_b = (const float*)d_in[4];
  const float* wk_w = (const float*)d_in[5];
  const float* wk_b = (const float*)d_in[6];
  const float* wv_w = (const float*)d_in[7];
  const float* wv_b = (const float*)d_in[8];
  const float* wo_w = (const float*)d_in[9];
  const float* wo_b = (const float*)d_in[10];
  const float* sinks = (const float*)d_in[11];
  float* out = (float*)d_out;

  char* ws = (char*)d_ws;
  __bf16* xb  = (__bf16*)(ws + 0);
  __bf16* wqb = (__bf16*)(ws + (8u << 20));
  __bf16* wkb = (__bf16*)(ws + (16u << 20));
  __bf16* wvb = (__bf16*)(ws + (18u << 20));
  __bf16* wob = (__bf16*)(ws + (20u << 20));
  __bf16* qb  = (__bf16*)(ws + (28u << 20));
  __bf16* kb  = (__bf16*)(ws + (36u << 20));
  __bf16* vt  = (__bf16*)(ws + (38u << 20));
  __bf16* ob  = (__bf16*)(ws + (40u << 20));

  auto cvt = [&](const float* s, __bf16* d, int n) {
    cvt_bf16_kernel<<<(n + 255) / 256, 256, 0, stream>>>(s, d, n);
  };
  cvt(x,    xb,  kS * kHID);
  cvt(wq_w, wqb, kNH * kHD * kHID);
  cvt(wk_w, wkb, kNKV * kHD * kHID);
  cvt(wv_w, wvb, kNKV * kHD * kHID);
  cvt(wo_w, wob, kHID * kNH * kHD);

  qkv_kernel<<<dim3(kS / 64, (kHID + 2 * kNKV * kHD) / 128), 128, 0, stream>>>(
      xb, wqb, wkb, wvb, wq_b, wk_b, wv_b, cosT, sinT, qb, kb, vt);

  attn_kernel<<<dim3(kNH, kS / 64), 128, 0, stream>>>(qb, kb, vt, sinks, ob);

  oproj_kernel<<<dim3(kS / 64, kHID / 128), 128, 0, stream>>>(ob, wob, wo_b, out);
}
